// WindowAttention_25194278159111
// MI455X (gfx1250) — matrix operand order repaired and checked by reading, not compiled, not run on hardware
//
#include <hip/hip_runtime.h>
#include <hip/hip_bf16.h>
#include <math.h>

typedef __attribute__((ext_vector_type(16))) _Float16 v16h;
typedef __attribute__((ext_vector_type(8)))  _Float16 v8h;
typedef __attribute__((ext_vector_type(8)))  float    v8f;
typedef __attribute__((ext_vector_type(4)))  float    v4f;
typedef int v4i __attribute__((vector_size(16)));    // matches async-LDS builtin pointee

#define NTOK   64
#define DIMSZ  128
#define NHEAD  4
#define HDIM   32
#define NWIN   192
// padded LDS strides (elements); all keep 16B alignment
#define XS 136   // x / normalized-O  [64][XS] f16
#define QS 40    // q,k,q2,k2  [4][64][QS] f16
#define VS 72    // vT         [4][32][VS] f16
#define PSTR 72  // probs      [4][64][PSTR] f16 (region doubles as f32 x-staging)

#if defined(__has_builtin)
#if __has_builtin(__builtin_amdgcn_global_load_async_to_lds_b128) && \
    __has_builtin(__builtin_amdgcn_s_wait_asynccnt)
#define HAVE_ASYNC_LDS 1
#endif
#endif

#if HAVE_ASYNC_LDS
typedef __attribute__((address_space(1))) v4i* g_v4i_ptr;   // global
typedef __attribute__((address_space(3))) v4i* l_v4i_ptr;   // LDS
#endif

// ---------------- helpers ----------------
__device__ __forceinline__ v16h ld2x8(const _Float16* p0, const _Float16* p1) {
    v8h lo = *(const v8h*)p0;
    v8h hh = *(const v8h*)p1;
    v16h r;
#pragma unroll
    for (int i = 0; i < 8; ++i) { r[i] = lo[i]; r[i + 8] = hh[i]; }
    return r;
}
// A operand (16x32 f16): lane<16 -> K koff+0..15; lane>=16 -> koff+16..31
__device__ __forceinline__ v16h afrag(const _Float16* base, int row, int stride,
                                      int koff, int hi16) {
    const _Float16* p = base + row * stride + koff + 16 * hi16;
    return ld2x8(p, p + 8);
}
// B operand: lane's column fixed, 16 contiguous K halves starting at p
__device__ __forceinline__ v16h bfrag(const _Float16* p) { return ld2x8(p, p + 8); }
// 16 consecutive f32 weights -> v16h (global, L2-resident)
__device__ __forceinline__ v16h wfrag(const float* p) {
    v16h r;
#pragma unroll
    for (int i = 0; i < 16; ++i) r[i] = (_Float16)p[i];
    return r;
}
// butterfly reductions across the 16-lane half-group (wave32, XOR stays in-group)
__device__ __forceinline__ float redsum16(float v) {
    v += __int_as_float(__builtin_amdgcn_ds_swizzle(__float_as_int(v), 0x041f)); // xor 1
    v += __int_as_float(__builtin_amdgcn_ds_swizzle(__float_as_int(v), 0x081f)); // xor 2
    v += __int_as_float(__builtin_amdgcn_ds_swizzle(__float_as_int(v), 0x101f)); // xor 4
    v += __int_as_float(__builtin_amdgcn_ds_swizzle(__float_as_int(v), 0x201f)); // xor 8
    return v;
}
__device__ __forceinline__ float redmax16(float v) {
    v = fmaxf(v, __int_as_float(__builtin_amdgcn_ds_swizzle(__float_as_int(v), 0x041f)));
    v = fmaxf(v, __int_as_float(__builtin_amdgcn_ds_swizzle(__float_as_int(v), 0x081f)));
    v = fmaxf(v, __int_as_float(__builtin_amdgcn_ds_swizzle(__float_as_int(v), 0x101f)));
    v = fmaxf(v, __int_as_float(__builtin_amdgcn_ds_swizzle(__float_as_int(v), 0x201f)));
    return v;
}

// ---------------- prep: lambda scalar + polar MLP bias ----------------
__global__ void wattn_prep_kernel(const float* __restrict__ Wp1, const float* __restrict__ bp1,
                                  const float* __restrict__ Wp2, const float* __restrict__ bp2,
                                  const float* __restrict__ lq1, const float* __restrict__ lk1,
                                  const float* __restrict__ lq2, const float* __restrict__ lk2,
                                  float* __restrict__ ws) {
    int id = blockIdx.x * blockDim.x + threadIdx.x;
    if (id == 0) {
        float d1 = 0.f, d2 = 0.f;
        for (int i = 0; i < 16; ++i) { d1 += lq1[i] * lk1[i]; d2 += lq2[i] * lk2[i]; }
        ws[0] = __expf(d1) - __expf(d2) + 0.2f;   // lambda_init = 0.2 (layer 0)
    }
    if (id < 4096) {
        int i = id >> 6, j = id & 63;
        float yi = (float)(i >> 3), xi = (float)(i & 7);
        float yj = (float)(j >> 3), xj = (float)(j & 7);
        float dx = xi - xj, dy = yj - yi;
        float r  = sqrtf(dx * dx + dy * dy + 1e-9f);
        float th = atan2f(dx, dy);
        float p0 = r * 0.10101525445522107f;                              // 1/sqrt(98)
        float p1 = (th + 3.14159265358979323846f) * 0.15915494309189535f; // /(2*pi)
        float o0 = bp2[0], o1 = bp2[1], o2 = bp2[2], o3 = bp2[3];
        for (int u = 0; u < 64; ++u) {
            float hpre = Wp1[2 * u] * p0 + Wp1[2 * u + 1] * p1 + bp1[u];
            float g = 0.5f * hpre * (1.0f + erff(hpre * 0.7071067811865476f)); // exact GELU
            o0 += Wp2[0 * 64 + u] * g;
            o1 += Wp2[1 * 64 + u] * g;
            o2 += Wp2[2 * 64 + u] * g;
            o3 += Wp2[3 * 64 + u] * g;
        }
        float* pb = ws + 64;
        pb[0 * 4096 + id] = o0;
        pb[1 * 4096 + id] = o1;
        pb[2 * 4096 + id] = o2;
        pb[3 * 4096 + id] = o3;
    }
}

// ---------------- fused differential window attention ----------------
__global__ __launch_bounds__(128)
void wattn_main_kernel(const float* __restrict__ xg, const float* __restrict__ maskg,
                       const float* __restrict__ Wqkv, const float* __restrict__ bqkv,
                       const float* __restrict__ Wqkv2, const float* __restrict__ bqkv2,
                       const float* __restrict__ subln, const float* __restrict__ Wproj,
                       const float* __restrict__ bproj, const float* __restrict__ ws,
                       float* __restrict__ outg) {
    __shared__ __align__(16) _Float16 xs[NTOK * XS];             // x f16 / final A operand
    __shared__ __align__(16) _Float16 qsm[NHEAD * NTOK * QS];
    __shared__ __align__(16) _Float16 ksm[NHEAD * NTOK * QS];
    __shared__ __align__(16) _Float16 q2s[NHEAD * NTOK * QS];
    __shared__ __align__(16) _Float16 k2s[NHEAD * NTOK * QS];
    __shared__ __align__(16) _Float16 vts[NHEAD * HDIM * VS];    // v^T [h][d][token]
    __shared__ __align__(16) _Float16 Psm[NHEAD * NTOK * PSTR];  // probs f16; phase-1 f32 x staging

    const int b    = blockIdx.x;
    const int tid  = threadIdx.x;
    const int wave = tid >> 5;
    const int lane = tid & 31;
    const int ln   = lane & 15;
    const int hi16 = lane >> 4;

    // weights live in L2 across all 3072 blocks: prefetch
    __builtin_prefetch(Wqkv  + tid * 4, 0, 3);
    __builtin_prefetch(Wqkv2 + tid * 4, 0, 3);
    __builtin_prefetch(Wproj + tid * 4, 0, 3);

    // ---- phase 1a: stage x (64x128 f32) into LDS, then convert to f16 ----
    {
        const float* xb = xg + (size_t)b * NTOK * DIMSZ;
#if HAVE_ASYNC_LDS
        float* stage = (float*)Psm;           // 32KB staging inside P region (dead until phase 2)
#pragma unroll
        for (int it = 0; it < 16; ++it) {
            int idx = (tid + 128 * it) * 4;   // float index of 16B chunk
            __builtin_amdgcn_global_load_async_to_lds_b128(
                (g_v4i_ptr)(xb + idx), (l_v4i_ptr)(stage + idx), 0, 0);
        }
        __builtin_amdgcn_s_wait_asynccnt(0);
        __syncthreads();
#pragma unroll
        for (int it = 0; it < 16; ++it) {
            int idx = tid + 128 * it;
            int row = idx >> 5;
            int col = (idx & 31) << 2;
            v4f xv = *(const v4f*)(stage + row * DIMSZ + col);
            _Float16* d = xs + row * XS + col;
            d[0] = (_Float16)xv[0]; d[1] = (_Float16)xv[1];
            d[2] = (_Float16)xv[2]; d[3] = (_Float16)xv[3];
        }
#else
#pragma unroll
        for (int it = 0; it < 16; ++it) {
            int idx = tid + 128 * it;
            int row = idx >> 5;
            int col = (idx & 31) << 2;
            v4f xv = *(const v4f*)(xb + row * DIMSZ + col);
            _Float16* d = xs + row * XS + col;
            d[0] = (_Float16)xv[0]; d[1] = (_Float16)xv[1];
            d[2] = (_Float16)xv[2]; d[3] = (_Float16)xv[3];
        }
#endif
    }
    __syncthreads();

    // ---- phase 1b: [q|k|v|q2|k2] = x @ Wcat^T ; 40 n-tiles of 16 channels ----
    for (int nt = wave; nt < 40; nt += 4) {
        const int c0 = nt * 16;
        const float* W;  const float* bias; int wrow;
        if (c0 < 384) { W = Wqkv;  bias = bqkv;  wrow = c0; }
        else          { W = Wqkv2; bias = bqkv2; wrow = c0 - 384; }
        const int ch = wrow + ln;
        const float bv = bias[ch];
        v16h bf[4];
#pragma unroll
        for (int kt = 0; kt < 4; ++kt)
            bf[kt] = wfrag(W + ch * DIMSZ + kt * 32 + 16 * hi16);

        const int head = (c0 & 127) >> 5;
        const int d    = (c0 & 31) + ln;
        _Float16* dst; int is_v = 0;
        if      (c0 < 128) dst = qsm;
        else if (c0 < 256) dst = ksm;
        else if (c0 < 384) { dst = vts; is_v = 1; }
        else if (c0 < 512) dst = q2s;
        else               dst = k2s;

        for (int mt = 0; mt < 4; ++mt) {
            v8f acc = {};
#pragma unroll
            for (int kt = 0; kt < 4; ++kt) {
                v16h a = afrag(xs, mt * 16 + ln, XS, kt * 32, hi16);
                acc = __builtin_amdgcn_wmma_f32_16x16x32_f16(
                          false, a, false, bf[kt], (short)0, acc, false, false);
            }
            if (!is_v) {
                _Float16* dh = dst + head * NTOK * QS;
#pragma unroll
                for (int r = 0; r < 8; ++r) {
                    int token = mt * 16 + r + 8 * hi16;
                    dh[token * QS + d] = (_Float16)(acc[r] + bv);
                }
            } else {
                _Float16* dh = vts + head * HDIM * VS + d * VS;
                int tb = mt * 16 + 8 * hi16;
                v8h pk;
#pragma unroll
                for (int r = 0; r < 8; ++r) pk[r] = (_Float16)(acc[r] + bv);
                *(v8h*)(dh + tb) = pk;
            }
        }
    }
    __syncthreads();
    // From here heads are wave-private: no cross-wave sync needed until phase 5.

    // ---- phase 2+3: logits, in-register softmax via ds_swizzle, P -> f16 LDS ----
    {
        const int h = wave;
        const float lam = ws[0];
        const float s1  = 0.17677669529663687f;   // 1/sqrt(32)
        const float s2c = lam * s1;
        const float* pbw = ws + 64 + h * 4096;
        const float* mk  = maskg + (size_t)(b % NWIN) * 4096;
        const _Float16* qh  = qsm + h * NTOK * QS;
        const _Float16* kh  = ksm + h * NTOK * QS;
        const _Float16* q2h = q2s + h * NTOK * QS;
        const _Float16* k2h = k2s + h * NTOK * QS;
        _Float16* Ph = Psm + h * NTOK * PSTR;

        // hoist all K / K2 B-fragments for this head (keys fully resident)
        v16h bk[4], bk2[4];
#pragma unroll
        for (int nt = 0; nt < 4; ++nt) {
            int key = nt * 16 + ln;
            bk[nt]  = bfrag(kh  + key * QS + 16 * hi16);
            bk2[nt] = bfrag(k2h + key * QS + 16 * hi16);
        }

        for (int mt = 0; mt < 4; ++mt) {
            v16h aq  = afrag(qh,  mt * 16 + ln, QS, 0, hi16);
            v16h aq2 = afrag(q2h, mt * 16 + ln, QS, 0, hi16);
            v8f sA[4], sB[4];
#pragma unroll
            for (int nt = 0; nt < 4; ++nt) {
                v8f za = {}, zb = {};
                sA[nt] = __builtin_amdgcn_wmma_f32_16x16x32_f16(
                             false, aq,  false, bk[nt],  (short)0, za, false, false);
                sB[nt] = __builtin_amdgcn_wmma_f32_16x16x32_f16(
                             false, aq2, false, bk2[nt], (short)0, zb, false, false);
            }
            const int ib = mt * 16 + 8 * hi16;
            float lg[4][8];
#pragma unroll
            for (int nt = 0; nt < 4; ++nt) {
                int j = nt * 16 + ln;
#pragma unroll
                for (int r = 0; r < 8; ++r) {
                    int i = ib + r;
                    lg[nt][r] = s1 * sA[nt][r] - s2c * sB[nt][r]
                              + pbw[i * 64 + j] + mk[i * 64 + j];
                }
            }
#pragma unroll
            for (int r = 0; r < 8; ++r) {
                float mx = fmaxf(fmaxf(lg[0][r], lg[1][r]), fmaxf(lg[2][r], lg[3][r]));
                mx = redmax16(mx);
                float e0 = __expf(lg[0][r] - mx);
                float e1 = __expf(lg[1][r] - mx);
                float e2 = __expf(lg[2][r] - mx);
                float e3 = __expf(lg[3][r] - mx);
                float inv = 1.0f / redsum16(e0 + e1 + e2 + e3);
                _Float16* Pr = Ph + (ib + r) * PSTR + ln;
                Pr[0]  = (_Float16)(e0 * inv);
                Pr[16] = (_Float16)(e1 * inv);
                Pr[32] = (_Float16)(e2 * inv);
                Pr[48] = (_Float16)(e3 * inv);
            }
        }
    }

    // ---- phase 4: O = P @ V, in-register RMSNorm, repack into xs ----
    {
        const int h = wave;
        const _Float16* Ph = Psm + h * NTOK * PSTR;
        const _Float16* Vh = vts + h * HDIM * VS;
        const float sl0 = subln[ln]      * 0.8f;   // *(1 - lambda_init)
        const float sl1 = subln[16 + ln] * 0.8f;
        v16h bvf[2][2];
#pragma unroll
        for (int nt = 0; nt < 2; ++nt)
#pragma unroll
            for (int kt = 0; kt < 2; ++kt)
                bvf[nt][kt] = bfrag(Vh + (nt * 16 + ln) * VS + kt * 32 + 16 * hi16);

        for (int mt = 0; mt < 4; ++mt) {
            v8f a0 = {}, a1 = {};
#pragma unroll
            for (int kt = 0; kt < 2; ++kt) {
                v16h a = afrag(Ph, mt * 16 + ln, PSTR, kt * 32, hi16);
                a0 = __builtin_amdgcn_wmma_f32_16x16x32_f16(
                         false, a, false, bvf[0][kt], (short)0, a0, false, false);
                a1 = __builtin_amdgcn_wmma_f32_16x16x32_f16(
                         false, a, false, bvf[1][kt], (short)0, a1, false, false);
            }
#pragma unroll
            for (int r = 0; r < 8; ++r) {
                float ss = redsum16(a0[r] * a0[r] + a1[r] * a1[r]);
                float rms = rsqrtf(ss * (1.0f / 32.0f) + 1e-9f);
                int token = mt * 16 + r + 8 * hi16;
                _Float16* dst = xs + token * XS + h * HDIM + ln;
                dst[0]  = (_Float16)(a0[r] * rms * sl0);
                dst[16] = (_Float16)(a1[r] * rms * sl1);
            }
        }
    }
    __syncthreads();

    // ---- phase 5: out = O @ Wproj^T + bproj -> f32 HBM ----
    {
        float* ob = outg + (size_t)b * NTOK * DIMSZ;
        for (int nt = wave; nt < 8; nt += 4) {
            const int ch = nt * 16 + ln;
            const float bv = bproj[ch];
            v16h bf[4];
#pragma unroll
            for (int kt = 0; kt < 4; ++kt)
                bf[kt] = wfrag(Wproj + ch * DIMSZ + kt * 32 + 16 * hi16);
            for (int mt = 0; mt < 4; ++mt) {
                v8f acc = {};
#pragma unroll
                for (int kt = 0; kt < 4; ++kt) {
                    v16h a = afrag(xs, mt * 16 + ln, XS, kt * 32, hi16);
                    acc = __builtin_amdgcn_wmma_f32_16x16x32_f16(
                              false, a, false, bf[kt], (short)0, acc, false, false);
                }
#pragma unroll
                for (int r = 0; r < 8; ++r) {
                    int token = mt * 16 + r + 8 * hi16;
                    ob[token * DIMSZ + ch] = acc[r] + bv;
                }
            }
        }
    }
}

extern "C" void kernel_launch(void* const* d_in, const int* in_sizes, int n_in,
                              void* d_out, int out_size, void* d_ws, size_t ws_size,
                              hipStream_t stream) {
    const float* x      = (const float*)d_in[0];
    const float* mask   = (const float*)d_in[1];
    const float* Wqkv   = (const float*)d_in[2];
    const float* bqkv   = (const float*)d_in[3];
    const float* Wqkv2  = (const float*)d_in[4];
    const float* bqkv2  = (const float*)d_in[5];
    const float* Wp1    = (const float*)d_in[6];
    const float* bp1    = (const float*)d_in[7];
    const float* Wp2    = (const float*)d_in[8];
    const float* bp2    = (const float*)d_in[9];
    const float* lq1    = (const float*)d_in[10];
    const float* lk1    = (const float*)d_in[11];
    const float* lq2    = (const float*)d_in[12];
    const float* lk2    = (const float*)d_in[13];
    const float* subln  = (const float*)d_in[14];
    const float* Wproj  = (const float*)d_in[15];
    const float* bproj  = (const float*)d_in[16];
    float* ws  = (float*)d_ws;
    float* out = (float*)d_out;

    wattn_prep_kernel<<<16, 256, 0, stream>>>(Wp1, bp1, Wp2, bp2, lq1, lk1, lq2, lk2, ws);
    wattn_main_kernel<<<3072, 128, 0, stream>>>(x, mask, Wqkv, bqkv, Wqkv2, bqkv2,
                                                subln, Wproj, bproj, ws, out);
}
